// EGNNLayer_8650064134312
// MI455X (gfx1250) — compile-verified, matrix operand order checked by reading, not run on hardware
//
#include <hip/hip_runtime.h>
#include <stdint.h>

// ---------------------------------------------------------------------------
// EGNN layer for MI455X (gfx1250, wave32, WMMA).
// Edge MLP / node MLP / coord MLP run on v_wmma_f32_16x16x32_bf16 (fp32 acc).
// One wave per 16-row tile; weights AND node features pre-converted to bf16
// in workspace so A/B fragments are straight b128 loads (no per-edge cvt).
// Scatter segment-sum via global_atomic_add_f32. LayerNorm via shfl_xor.
// ---------------------------------------------------------------------------

typedef __attribute__((ext_vector_type(16))) __bf16 v16bf;
typedef __attribute__((ext_vector_type(4)))  __bf16 v4bf;
typedef __attribute__((ext_vector_type(8)))  float  v8f;

#define DD   128          // node feature dim
#define HH   128          // hidden dim
#define K1   288          // 261 padded to 9*32
#define LDSR 136          // padded LDS row stride (bf16 elems) to spread banks

__device__ __forceinline__ int kpair(int v, int hi) {
    // 16-bit A/B fragment K-pair start within a 32-wide K tile (ISA 7.12.2)
    return (hi ? 8 : 0) + (v < 4 ? (v << 1) : 16 + ((v - 4) << 1));
}

__device__ __forceinline__ float silu(float v) {
    return v * __builtin_amdgcn_rcpf(1.0f + __expf(-v));
}

// A fragment from a contiguous fp32 row (convert to bf16 on the fly)
__device__ __forceinline__ v16bf fragA_row(const float* __restrict__ row,
                                           int koff, int hi) {
    v16bf a;
#pragma unroll
    for (int v = 0; v < 8; ++v) {
        int k = koff + kpair(v, hi);
        a[2 * v]     = (__bf16)row[k];
        a[2 * v + 1] = (__bf16)row[k + 1];
    }
    return a;
}

// A/B fragment from a contiguous bf16 row (weights / hbf / LDS tile row)
__device__ __forceinline__ v16bf frag_bf16(const __bf16* __restrict__ wrow,
                                           int hi) {
    v16bf b;
#pragma unroll
    for (int v = 0; v < 8; ++v) {
        int k = kpair(v, hi);
        b[2 * v]     = wrow[k];
        b[2 * v + 1] = wrow[k + 1];
    }
    return b;
}

__device__ __forceinline__ float edge_tail(int k, float dsq,
                                           const float* __restrict__ ea) {
    if (k == 256) return dsq;
    if (k > 256 && k < 261) return ea[k - 257];
    return 0.0f;
}

// ------------------------- weight prep (fp32 -> bf16, transposed) -----------
#define S_W1T (128 * K1)
#define S_W2T (128 * 128)
#define S_N1T (128 * 256)
#define S_N2T (128 * 128)
#define S_C1T (64 * 128)
#define S_TOT (S_W1T + S_W2T + S_N1T + S_N2T + S_C1T)

__global__ void egnn_prep_kernel(const float* __restrict__ ew1,
                                 const float* __restrict__ ew2,
                                 const float* __restrict__ nw1,
                                 const float* __restrict__ nw2,
                                 const float* __restrict__ cw1,
                                 __bf16* __restrict__ W1t, __bf16* __restrict__ W2t,
                                 __bf16* __restrict__ N1t, __bf16* __restrict__ N2t,
                                 __bf16* __restrict__ C1t) {
    int id = blockIdx.x * 256 + threadIdx.x;
    if (id < S_W1T) { int n = id / K1, k = id % K1;
        W1t[id] = (k < 261) ? (__bf16)ew1[k * 128 + n] : (__bf16)0.0f; return; }
    id -= S_W1T;
    if (id < S_W2T) { int n = id / 128, k = id % 128;
        W2t[id] = (__bf16)ew2[k * 128 + n]; return; }
    id -= S_W2T;
    if (id < S_N1T) { int n = id / 256, k = id % 256;
        N1t[id] = (__bf16)nw1[k * 128 + n]; return; }
    id -= S_N1T;
    if (id < S_N2T) { int n = id / 128, k = id % 128;
        N2t[id] = (__bf16)nw2[k * 128 + n]; return; }
    id -= S_N2T;
    if (id < S_C1T) { int n = id / 128, k = id % 128;
        C1t[id] = (__bf16)cw1[k * 64 + n]; return; }
}

// node features fp32 -> bf16 copy (vectorized: float4 in, 4x bf16 out)
__global__ void egnn_h2bf_kernel(const float* __restrict__ h,
                                 __bf16* __restrict__ hbf, long long total4) {
    long long i = blockIdx.x * (long long)blockDim.x + threadIdx.x;
    if (i >= total4) return;
    float4 f = *(const float4*)(h + i * 4);
    v4bf o;
    o[0] = (__bf16)f.x; o[1] = (__bf16)f.y;
    o[2] = (__bf16)f.z; o[3] = (__bf16)f.w;
    *(v4bf*)(hbf + i * 4) = o;
}

// ------------------------- edge kernel --------------------------------------
// One wave per 16-edge tile; 8 waves (256 threads) per block.
__global__ __launch_bounds__(256)
void egnn_edge_kernel(const __bf16* __restrict__ hbf, const float* __restrict__ x,
                      const long long* __restrict__ ei,
                      const float* __restrict__ ea,
                      const float* __restrict__ eb1, const float* __restrict__ eb2,
                      const float* __restrict__ cb1, const float* __restrict__ cw2,
                      const float* __restrict__ cb2,
                      const __bf16* __restrict__ W1t,
                      const __bf16* __restrict__ W2t,
                      const __bf16* __restrict__ C1t,
                      float* __restrict__ agg, float* __restrict__ delta,
                      float* __restrict__ deg, int E) {
    __shared__ __bf16 stile[8][16][LDSR];
    __shared__ float  swv[8][16];

    const int lane = threadIdx.x & 31;
    const int wav  = threadIdx.x >> 5;
    const int tile = blockIdx.x * 8 + wav;
    if (tile * 16 >= E) return;

    const int m  = lane & 15;      // this lane's edge row / output column n0
    const int hi = lane >> 4;
    const int e  = tile * 16 + m;  // E is a multiple of 16

    const long long srcn = ei[e];
    const long long dstn = ei[(long long)E + e];
    const float* xs = x + srcn * 3;
    const float* xd = x + dstn * 3;
    float rx = xd[0] - xs[0], ry = xd[1] - xs[1], rz = xd[2] - xs[2];
    float dsq = fmaxf(rx * rx + ry * ry + rz * rz, 1e-8f);
    float dist = sqrtf(dsq);
    const __bf16* hdst  = hbf + dstn * DD;
    const __bf16* hsrc  = hbf + srcn * DD;
    const float*  eattr = ea + (long long)e * 4;

    // ---- MLP1: edge_in(16x288) @ ew1(288x128) + eb1 ----
    v8f acc[8];
#pragma unroll
    for (int nt = 0; nt < 8; ++nt) {
        float b = eb1[nt * 16 + m];
#pragma unroll
        for (int r = 0; r < 8; ++r) acc[nt][r] = b;
    }
#pragma unroll
    for (int kt = 0; kt < 9; ++kt) {
        v16bf a;
        if (kt < 4)      a = frag_bf16(hdst + kt * 32, hi);
        else if (kt < 8) a = frag_bf16(hsrc + (kt - 4) * 32, hi);
        else {
#pragma unroll
            for (int v = 0; v < 8; ++v) {
                int k = 256 + kpair(v, hi);
                a[2 * v]     = (__bf16)edge_tail(k, dsq, eattr);
                a[2 * v + 1] = (__bf16)edge_tail(k + 1, dsq, eattr);
            }
        }
#pragma unroll
        for (int nt = 0; nt < 8; ++nt) {
            v16bf b = frag_bf16(W1t + (nt * 16 + m) * K1 + kt * 32, hi);
            acc[nt] = __builtin_amdgcn_wmma_f32_16x16x32_bf16(
                false, a, false, b, (short)0, acc[nt], false, false);
        }
    }
    // silu -> stage to LDS in A-layout-friendly row-major form
#pragma unroll
    for (int nt = 0; nt < 8; ++nt)
#pragma unroll
        for (int r = 0; r < 8; ++r)
            stile[wav][r + 8 * hi][nt * 16 + m] = (__bf16)silu(acc[nt][r]);
    __builtin_amdgcn_wave_barrier();

    // ---- MLP2: t1(16x128) @ ew2(128x128) + eb2 ----
    v8f acc2[8];
#pragma unroll
    for (int nt = 0; nt < 8; ++nt) {
        float b = eb2[nt * 16 + m];
#pragma unroll
        for (int r = 0; r < 8; ++r) acc2[nt][r] = b;
    }
#pragma unroll
    for (int kt = 0; kt < 4; ++kt) {
        v16bf a = frag_bf16(&stile[wav][m][kt * 32], hi);
#pragma unroll
        for (int nt = 0; nt < 8; ++nt) {
            v16bf b = frag_bf16(W2t + (nt * 16 + m) * 128 + kt * 32, hi);
            acc2[nt] = __builtin_amdgcn_wmma_f32_16x16x32_bf16(
                false, a, false, b, (short)0, acc2[nt], false, false);
        }
    }
    // silu -> msg; restage to LDS and scatter-add to agg[dst]
    __builtin_amdgcn_wave_barrier();
#pragma unroll
    for (int nt = 0; nt < 8; ++nt)
#pragma unroll
        for (int r = 0; r < 8; ++r) {
            float v = silu(acc2[nt][r]);
            acc2[nt][r] = v;
            stile[wav][r + 8 * hi][nt * 16 + m] = (__bf16)v;
        }
    __builtin_amdgcn_wave_barrier();

    int dst32 = (int)dstn;
#pragma unroll
    for (int r = 0; r < 8; ++r) {
        int row = r + 8 * hi;
        int dr  = __shfl(dst32, row, 32);
        float* dp = agg + (long long)dr * HH + m;
#pragma unroll
        for (int nt = 0; nt < 8; ++nt)
            atomicAdd(dp + nt * 16, acc2[nt][r]);
    }

    // ---- coord MLP: msg(16x128) @ cw1(128x64) -> silu -> @ cw2(64x1) ----
    float wsum[8];
#pragma unroll
    for (int r = 0; r < 8; ++r) wsum[r] = 0.0f;
#pragma unroll
    for (int nt = 0; nt < 4; ++nt) {
        int n = nt * 16 + m;
        v8f accc;
        float b0 = cb1[n];
#pragma unroll
        for (int r = 0; r < 8; ++r) accc[r] = b0;
#pragma unroll
        for (int kt = 0; kt < 4; ++kt) {
            v16bf a = frag_bf16(&stile[wav][m][kt * 32], hi);
            v16bf b = frag_bf16(C1t + n * 128 + kt * 32, hi);
            accc = __builtin_amdgcn_wmma_f32_16x16x32_bf16(
                false, a, false, b, (short)0, accc, false, false);
        }
        float w2 = cw2[n];
#pragma unroll
        for (int r = 0; r < 8; ++r) wsum[r] += silu(accc[r]) * w2;
    }
    // 16-lane butterfly within each half-wave: sum over the 64 columns
#pragma unroll
    for (int off = 1; off < 16; off <<= 1)
#pragma unroll
        for (int r = 0; r < 8; ++r) wsum[r] += __shfl_xor(wsum[r], off, 32);
    if (m == 0) {                          // lane 0 -> rows 0..7, lane 16 -> 8..15
#pragma unroll
        for (int r = 0; r < 8; ++r) swv[wav][r + 8 * hi] = wsum[r];
    }
    __builtin_amdgcn_wave_barrier();

    if (hi == 0) {                         // one lane per edge does coord scatter
        float wv   = swv[wav][m] + cb2[0];
        float invd = wv / (dist + 1.0f);
        atomicAdd(delta + (long long)dst32 * 3 + 0, rx * invd);
        atomicAdd(delta + (long long)dst32 * 3 + 1, ry * invd);
        atomicAdd(delta + (long long)dst32 * 3 + 2, rz * invd);
        atomicAdd(deg + dst32, 1.0f);
    }
}

// ------------------------- node kernel --------------------------------------
__global__ __launch_bounds__(256)
void egnn_node_kernel(const float* __restrict__ h, const __bf16* __restrict__ hbf,
                      const float* __restrict__ agg,
                      const float* __restrict__ nb1, const float* __restrict__ nb2,
                      const __bf16* __restrict__ N1t, const __bf16* __restrict__ N2t,
                      const float* __restrict__ ln_g, const float* __restrict__ ln_b,
                      float* __restrict__ hout, int N) {
    __shared__ __bf16 stile[8][16][LDSR];

    const int lane = threadIdx.x & 31;
    const int wav  = threadIdx.x >> 5;
    const int tile = blockIdx.x * 8 + wav;
    if (tile * 16 >= N) return;

    const int m  = lane & 15;
    const int hi = lane >> 4;
    const long long node = (long long)tile * 16 + m;   // N multiple of 16

    const __bf16* hrow = hbf + node * DD;
    const float*  arow = agg + node * HH;

    // ---- MLP1: [h|agg](16x256) @ nw1(256x128) + nb1 ----
    v8f acc[8];
#pragma unroll
    for (int nt = 0; nt < 8; ++nt) {
        float b = nb1[nt * 16 + m];
#pragma unroll
        for (int r = 0; r < 8; ++r) acc[nt][r] = b;
    }
#pragma unroll
    for (int kt = 0; kt < 8; ++kt) {
        v16bf a = (kt < 4) ? frag_bf16(hrow + kt * 32, hi)
                           : fragA_row(arow, kt * 32 - 128, hi);
#pragma unroll
        for (int nt = 0; nt < 8; ++nt) {
            v16bf b = frag_bf16(N1t + (nt * 16 + m) * 256 + kt * 32, hi);
            acc[nt] = __builtin_amdgcn_wmma_f32_16x16x32_bf16(
                false, a, false, b, (short)0, acc[nt], false, false);
        }
    }
#pragma unroll
    for (int nt = 0; nt < 8; ++nt)
#pragma unroll
        for (int r = 0; r < 8; ++r)
            stile[wav][r + 8 * hi][nt * 16 + m] = (__bf16)silu(acc[nt][r]);
    __builtin_amdgcn_wave_barrier();

    // ---- MLP2: t1(16x128) @ nw2(128x128) + nb2 ----
    v8f acc2[8];
#pragma unroll
    for (int nt = 0; nt < 8; ++nt) {
        float b = nb2[nt * 16 + m];
#pragma unroll
        for (int r = 0; r < 8; ++r) acc2[nt][r] = b;
    }
#pragma unroll
    for (int kt = 0; kt < 4; ++kt) {
        v16bf a = frag_bf16(&stile[wav][m][kt * 32], hi);
#pragma unroll
        for (int nt = 0; nt < 8; ++nt) {
            v16bf b = frag_bf16(N2t + (nt * 16 + m) * 128 + kt * 32, hi);
            acc2[nt] = __builtin_amdgcn_wmma_f32_16x16x32_bf16(
                false, a, false, b, (short)0, acc2[nt], false, false);
        }
    }

    // residual + LayerNorm (per row: 8 local cols x 16 lanes = 128)
    const float* hbase = h + (long long)tile * 16 * DD;
#pragma unroll
    for (int nt = 0; nt < 8; ++nt)
#pragma unroll
        for (int r = 0; r < 8; ++r)
            acc2[nt][r] += hbase[(r + 8 * hi) * DD + nt * 16 + m];

    float s[8];
#pragma unroll
    for (int r = 0; r < 8; ++r) {
        s[r] = 0.0f;
#pragma unroll
        for (int nt = 0; nt < 8; ++nt) s[r] += acc2[nt][r];
    }
#pragma unroll
    for (int off = 1; off < 16; off <<= 1)
#pragma unroll
        for (int r = 0; r < 8; ++r) s[r] += __shfl_xor(s[r], off, 32);

    float mu[8], q[8];
#pragma unroll
    for (int r = 0; r < 8; ++r) {
        mu[r] = s[r] * (1.0f / 128.0f);
        q[r]  = 0.0f;
#pragma unroll
        for (int nt = 0; nt < 8; ++nt) {
            float d = acc2[nt][r] - mu[r];
            q[r] += d * d;
        }
    }
#pragma unroll
    for (int off = 1; off < 16; off <<= 1)
#pragma unroll
        for (int r = 0; r < 8; ++r) q[r] += __shfl_xor(q[r], off, 32);

    float rstd[8];
#pragma unroll
    for (int r = 0; r < 8; ++r)
        rstd[r] = rsqrtf(q[r] * (1.0f / 128.0f) + 1e-5f);

    float* obase = hout + (long long)tile * 16 * DD;
#pragma unroll
    for (int nt = 0; nt < 8; ++nt) {
        float g = ln_g[nt * 16 + m];
        float b = ln_b[nt * 16 + m];
#pragma unroll
        for (int r = 0; r < 8; ++r)
            obase[(r + 8 * hi) * DD + nt * 16 + m] =
                (acc2[nt][r] - mu[r]) * rstd[r] * g + b;
    }
}

// ------------------------- coord finalize -----------------------------------
__global__ void egnn_coord_kernel(const float* __restrict__ x,
                                  const float* __restrict__ delta,
                                  const float* __restrict__ deg,
                                  const float* __restrict__ logit,
                                  float* __restrict__ xout, int N) {
    int i = blockIdx.x * blockDim.x + threadIdx.x;
    if (i >= N) return;
    float sc = 1.0f / (1.0f + __expf(-logit[0]));
    float dg = fmaxf(deg[i], 1.0f);
    float inv = sc / dg;
#pragma unroll
    for (int c = 0; c < 3; ++c)
        xout[(long long)i * 3 + c] = x[(long long)i * 3 + c] +
                                     delta[(long long)i * 3 + c] * inv;
}

// ------------------------- launch -------------------------------------------
extern "C" void kernel_launch(void* const* d_in, const int* in_sizes, int n_in,
                              void* d_out, int out_size, void* d_ws, size_t ws_size,
                              hipStream_t stream) {
    const float*     h     = (const float*)d_in[0];
    const float*     x     = (const float*)d_in[1];
    const long long* ei    = (const long long*)d_in[2];
    const float*     ea    = (const float*)d_in[3];
    const float*     ew1   = (const float*)d_in[4];
    const float*     eb1   = (const float*)d_in[5];
    const float*     ew2   = (const float*)d_in[6];
    const float*     eb2   = (const float*)d_in[7];
    const float*     nw1   = (const float*)d_in[8];
    const float*     nb1   = (const float*)d_in[9];
    const float*     nw2   = (const float*)d_in[10];
    const float*     nb2   = (const float*)d_in[11];
    const float*     cw1   = (const float*)d_in[12];
    const float*     cb1   = (const float*)d_in[13];
    const float*     cw2   = (const float*)d_in[14];
    const float*     cb2   = (const float*)d_in[15];
    const float*     logit = (const float*)d_in[16];
    const float*     ln_g  = (const float*)d_in[17];
    const float*     ln_b  = (const float*)d_in[18];

    const int N = in_sizes[0] / DD;
    const int E = in_sizes[2] / 2;

    // workspace layout (bytes)
    char* ws = (char*)d_ws;
    size_t off = 0;
    __bf16* W1t = (__bf16*)(ws + off); off += (size_t)S_W1T * 2;
    __bf16* W2t = (__bf16*)(ws + off); off += (size_t)S_W2T * 2;
    __bf16* N1t = (__bf16*)(ws + off); off += (size_t)S_N1T * 2;
    __bf16* N2t = (__bf16*)(ws + off); off += (size_t)S_N2T * 2;
    __bf16* C1t = (__bf16*)(ws + off); off += (size_t)S_C1T * 2;
    off = (off + 255) & ~(size_t)255;
    __bf16* hbf = (__bf16*)(ws + off); off += (size_t)N * DD * 2;
    off = (off + 255) & ~(size_t)255;
    float* agg   = (float*)(ws + off); off += (size_t)N * HH * 4;
    float* delta = (float*)(ws + off); off += (size_t)N * 3 * 4;
    float* deg   = (float*)(ws + off); off += (size_t)N * 4;

    float* hout = (float*)d_out;
    float* xout = hout + (size_t)N * DD;

    // 1) weights -> bf16 transposed; h -> bf16 copy
    egnn_prep_kernel<<<(S_TOT + 255) / 256, 256, 0, stream>>>(
        ew1, ew2, nw1, nw2, cw1, W1t, W2t, N1t, N2t, C1t);
    long long total4 = (long long)N * DD / 4;
    egnn_h2bf_kernel<<<(int)((total4 + 255) / 256), 256, 0, stream>>>(
        h, hbf, total4);

    // 2) zero accumulators (agg, delta, deg are contiguous)
    hipMemsetAsync(agg, 0, (size_t)N * (HH + 4) * 4, stream);

    // 3) edge MLP + msg aggregation + coord weights
    int etiles = (E + 15) / 16;
    egnn_edge_kernel<<<(etiles + 7) / 8, 256, 0, stream>>>(
        hbf, x, ei, ea, eb1, eb2, cb1, cw2, cb2,
        W1t, W2t, C1t, agg, delta, deg, E);

    // 4) node MLP + residual + LayerNorm
    int ntiles = (N + 15) / 16;
    egnn_node_kernel<<<(ntiles + 7) / 8, 256, 0, stream>>>(
        h, hbf, agg, nb1, nb2, N1t, N2t, ln_g, ln_b, hout, N);

    // 5) coordinate update
    egnn_coord_kernel<<<(N + 255) / 256, 256, 0, stream>>>(
        x, delta, deg, logit, xout, N);
}
